// SAModule_68719477563
// MI455X (gfx1250) — matrix operand order repaired and checked by reading, not compiled, not run on hardware
//
#include <hip/hip_runtime.h>

// ---------------- problem constants (match reference) ----------------
constexpr int NB   = 8;     // batches
constexpr int NPTS = 4096;  // points per cloud
constexpr int CIN  = 32;    // input channels
constexpr int MCTR = 1024;  // centroids per cloud (N * 0.25)
constexpr int KNB  = 64;    // max neighbors
constexpr float R2V = 0.04f;  // r^2 = 0.2^2
constexpr float NEGV = -1e10f;
constexpr int CAP = 1024;   // in-ball candidate capacity

// ---------------- WMMA helper types ----------------
typedef __attribute__((ext_vector_type(16))) _Float16 v16h;
typedef __attribute__((ext_vector_type(8)))  _Float16 v8h;
typedef __attribute__((ext_vector_type(8)))  float    v8f;

// A fragment (16x32 f16) from row-major LDS tile with row stride 64 halves.
// `tile` points at (row0, k0). Lane layout per CDNA5 ISA:
//   lanes 0-15: M=lane, K = {0..7, 16..23}; lanes 16-31: M=lane-16, K = {8..15, 24..31}
__device__ __forceinline__ v16h frag_a(const _Float16* tile, int lane) {
  const _Float16* r = tile + (lane & 15) * 64 + (lane >> 4) * 8;
  v8h lo = *(const v8h*)(r);        // K = 8*lh + 0..7
  v8h hi = *(const v8h*)(r + 16);   // K = 16 + 8*lh + 0..7
  return __builtin_shufflevector(lo, hi, 0,1,2,3,4,5,6,7,8,9,10,11,12,13,14,15);
}

// B fragment (32x16 f16) from transposed weights Wt[n][k] (k contiguous, stride 64).
// lanes 0-15: N=lane, K = {0..7, 16..23}; lanes 16-31: N=lane-16, K = {8..15, 24..31}
__device__ __forceinline__ v16h frag_b(const _Float16* wt, int col0, int k0, int lane) {
  const _Float16* r = wt + (size_t)(col0 + (lane & 15)) * 64 + k0 + (lane >> 4) * 8;
  v8h lo = *(const v8h*)(r);
  v8h hi = *(const v8h*)(r + 16);
  return __builtin_shufflevector(lo, hi, 0,1,2,3,4,5,6,7,8,9,10,11,12,13,14,15);
}

// ---------------- Kernel 1: farthest point sampling ----------------
__global__ __launch_bounds__(256) void fps_kernel(const float* __restrict__ pos,
                                                  int* __restrict__ idx,
                                                  float* __restrict__ pos_out,
                                                  int* __restrict__ batch_out) {
  const int b = blockIdx.x;
  const int tid = threadIdx.x;
  __shared__ float sx[NPTS], sy[NPTS], sz[NPTS];
  __shared__ int   sIdx[MCTR];
  __shared__ float bcast[3];
  __shared__ float redV[8];
  __shared__ int   redI[8];

  const float* p = pos + (size_t)b * NPTS * 3;
  for (int i = tid; i < NPTS; i += 256) {
    sx[i] = p[i * 3 + 0];
    sy[i] = p[i * 3 + 1];
    sz[i] = p[i * 3 + 2];
  }
  float dmin[NPTS / 256];
#pragma unroll
  for (int t = 0; t < NPTS / 256; t++) dmin[t] = 1e10f;
  if (tid == 0) { idx[b * MCTR + 0] = 0; sIdx[0] = 0; }
  __syncthreads();

  float lx = sx[0], ly = sy[0], lz = sz[0];
  for (int m = 1; m < MCTR; m++) {
    float bestV = -1.0f; int bestI = 0;
#pragma unroll
    for (int t = 0; t < NPTS / 256; t++) {
      int i = tid + t * 256;
      float dx = sx[i] - lx, dy = sy[i] - ly, dz = sz[i] - lz;
      float d = dx * dx + dy * dy + dz * dz;
      float dm = fminf(dmin[t], d);
      dmin[t] = dm;
      if (dm > bestV) { bestV = dm; bestI = i; }   // ascending i -> first max kept
    }
    // wave32 reduce (argmax, lowest index on ties)
    for (int off = 16; off > 0; off >>= 1) {
      float ov = __shfl_down(bestV, off);
      int   oi = __shfl_down(bestI, off);
      if (ov > bestV || (ov == bestV && oi < bestI)) { bestV = ov; bestI = oi; }
    }
    if ((tid & 31) == 0) { redV[tid >> 5] = bestV; redI[tid >> 5] = bestI; }
    __syncthreads();
    if (tid == 0) {
      float bv = redV[0]; int bi = redI[0];
      for (int w = 1; w < 8; w++)
        if (redV[w] > bv || (redV[w] == bv && redI[w] < bi)) { bv = redV[w]; bi = redI[w]; }
      idx[b * MCTR + m] = bi;
      sIdx[m] = bi;
      bcast[0] = sx[bi]; bcast[1] = sy[bi]; bcast[2] = sz[bi];
    }
    __syncthreads();
    lx = bcast[0]; ly = bcast[1]; lz = bcast[2];
  }
  __syncthreads();
  for (int m = tid; m < MCTR; m += 256) {
    int gi = sIdx[m];
    size_t o = (size_t)(b * MCTR + m);
    pos_out[o * 3 + 0] = sx[gi];
    pos_out[o * 3 + 1] = sy[gi];
    pos_out[o * 3 + 2] = sz[gi];
    batch_out[o] = b;
  }
}

// ---------------- Kernel 2: ball query (= top-K masked by r) ----------------
__global__ __launch_bounds__(256) void ball_kernel(const float* __restrict__ pos,
                                                   const int* __restrict__ idx,
                                                   int* __restrict__ nbr,
                                                   int* __restrict__ cnt) {
  const int c = blockIdx.x;        // centroid id in [0, NB*MCTR)
  const int b = c / MCTR;
  const int tid = threadIdx.x;
  __shared__ float cd2[CAP];
  __shared__ int   cid[CAP];
  __shared__ int   scnt;
  __shared__ float redV[8];
  __shared__ int   redI[8];

  const float* p = pos + (size_t)b * NPTS * 3;
  const int gi = idx[c];
  const float cx = p[gi * 3 + 0], cy = p[gi * 3 + 1], cz = p[gi * 3 + 2];
  if (tid == 0) scnt = 0;
  __syncthreads();

  for (int i = tid; i < NPTS; i += 256) {
    float dx = cx - p[i * 3 + 0];
    float dy = cy - p[i * 3 + 1];
    float dz = cz - p[i * 3 + 2];
    float d = dx * dx + dy * dy + dz * dz;
    if (d <= R2V) {
      int s = atomicAdd(&scnt, 1);
      if (s < CAP) { cd2[s] = d; cid[s] = i; }
    }
  }
  __syncthreads();
  int count = scnt; if (count > CAP) count = CAP;

  if (count <= KNB) {
    for (int j = tid; j < count; j += 256) nbr[c * KNB + j] = cid[j];
    for (int j = count + tid; j < KNB; j += 256) nbr[c * KNB + j] = 0;  // masked later
  } else {
    // select the KNB nearest among the in-ball candidates
    for (int s = 0; s < KNB; s++) {
      float bv = 1e30f; int bi = -1;
      for (int j = tid; j < count; j += 256) {
        float v = cd2[j];
        if (v < bv) { bv = v; bi = j; }
      }
      for (int off = 16; off > 0; off >>= 1) {
        float ov = __shfl_down(bv, off);
        int   oi = __shfl_down(bi, off);
        if (ov < bv) { bv = ov; bi = oi; }
      }
      if ((tid & 31) == 0) { redV[tid >> 5] = bv; redI[tid >> 5] = bi; }
      __syncthreads();
      if (tid == 0) {
        float mv = redV[0]; int mi = redI[0];
        for (int w = 1; w < 8; w++) if (redV[w] < mv) { mv = redV[w]; mi = redI[w]; }
        nbr[c * KNB + s] = cid[mi];
        cd2[mi] = 1e30f;  // mark used
      }
      __syncthreads();
    }
    count = KNB;
  }
  if (tid == 0) cnt[c] = count;
}

// ---------------- Kernel 3: gather + MLP (WMMA) + masked max ----------------
__global__ __launch_bounds__(128) void mlp_kernel(
    const float* __restrict__ x, const float* __restrict__ pos,
    const int* __restrict__ idx, const int* __restrict__ nbr,
    const int* __restrict__ cnt,
    const float* __restrict__ W1, const float* __restrict__ b1,
    const float* __restrict__ W2, const float* __restrict__ b2,
    const float* __restrict__ W3, const float* __restrict__ b3,
    float* __restrict__ xout) {
  const int c   = blockIdx.x;
  const int bb  = c / MCTR;
  const int tid = threadIdx.x;
  const int lane = tid & 31;
  const int wv   = tid >> 5;    // wave id 0..3 -> row tile
  const int lh   = lane >> 4;
  const int ln   = lane & 15;

  __shared__ __align__(16) _Float16 Xs [64 * 64];   // [nbr slot][k]  (35 used)
  __shared__ __align__(16) _Float16 Wt1[64 * 64];   // [n][k] transposed
  __shared__ __align__(16) _Float16 Wt2[64 * 64];
  __shared__ __align__(16) _Float16 Wt3[128 * 64];
  __shared__ __align__(16) _Float16 H1 [64 * 64];
  __shared__ __align__(16) _Float16 H2 [64 * 64];
  __shared__ float b1s[64], b2s[64], b3s[128];
  __shared__ float pmax[4][128];
  __shared__ int   snbr[64];
  __shared__ float cc[3];

  const float* p = pos + (size_t)bb * NPTS * 3;
  const int vcnt = cnt[c];

  if (tid < 64) snbr[tid] = nbr[c * KNB + tid];
  if (tid == 0) {
    int gi = idx[c];
    cc[0] = p[gi * 3 + 0]; cc[1] = p[gi * 3 + 1]; cc[2] = p[gi * 3 + 2];
  }
  if (tid < 64)  { b1s[tid] = b1[tid]; b2s[tid] = b2[tid]; }
  b3s[tid] = b3[tid];

  for (int e = tid; e < 64 * 64; e += 128) {
    int n = e >> 6, k = e & 63;
    Wt1[e] = (_Float16)((k < CIN + 3) ? W1[k * 64 + n] : 0.0f);
    Wt2[e] = (_Float16)(W2[k * 64 + n]);
  }
  for (int e = tid; e < 128 * 64; e += 128) {
    int n = e >> 6, k = e & 63;
    Wt3[e] = (_Float16)(W3[k * 128 + n]);
  }
  __syncthreads();

  // gather input rows: [x_j (32) | pos_j - pos_i (3) | zero pad]
  for (int e = tid; e < 64 * 64; e += 128) {
    int j = e >> 6, f = e & 63;
    int g = snbr[j];
    float v = 0.0f;
    if (f < CIN)          v = x[((size_t)bb * NPTS + g) * CIN + f];
    else if (f < CIN + 3) v = p[g * 3 + (f - CIN)] - cc[f - CIN];
    Xs[e] = (_Float16)v;
  }
  __syncthreads();

  const v8f vz = {0, 0, 0, 0, 0, 0, 0, 0};

  // ---- layer 1: H1 = relu(Xs @ W1 + b1)   (64x64, K=64 padded) ----
  v8f acc1[4];
#pragma unroll
  for (int t = 0; t < 4; t++) acc1[t] = vz;
  for (int k0 = 0; k0 < 64; k0 += 32) {
    v16h a = frag_a(Xs + wv * 16 * 64 + k0, lane);
#pragma unroll
    for (int tn = 0; tn < 4; tn++) {
      v16h bf = frag_b(Wt1, tn * 16, k0, lane);
      acc1[tn] = __builtin_amdgcn_wmma_f32_16x16x32_f16(false, a, false, bf,
                                                        (short)0, acc1[tn], false, false);
    }
  }
#pragma unroll
  for (int tn = 0; tn < 4; tn++) {
    int col = tn * 16 + ln;
    float bv = b1s[col];
#pragma unroll
    for (int e = 0; e < 8; e++) {
      float v = acc1[tn][e] + bv;
      H1[(wv * 16 + lh * 8 + e) * 64 + col] = (_Float16)fmaxf(v, 0.0f);
    }
  }
  __syncthreads();

  // ---- layer 2: H2 = relu(H1 @ W2 + b2) ----
  v8f acc2[4];
#pragma unroll
  for (int t = 0; t < 4; t++) acc2[t] = vz;
  for (int k0 = 0; k0 < 64; k0 += 32) {
    v16h a = frag_a(H1 + wv * 16 * 64 + k0, lane);
#pragma unroll
    for (int tn = 0; tn < 4; tn++) {
      v16h bf = frag_b(Wt2, tn * 16, k0, lane);
      acc2[tn] = __builtin_amdgcn_wmma_f32_16x16x32_f16(false, a, false, bf,
                                                        (short)0, acc2[tn], false, false);
    }
  }
#pragma unroll
  for (int tn = 0; tn < 4; tn++) {
    int col = tn * 16 + ln;
    float bv = b2s[col];
#pragma unroll
    for (int e = 0; e < 8; e++) {
      float v = acc2[tn][e] + bv;
      H2[(wv * 16 + lh * 8 + e) * 64 + col] = (_Float16)fmaxf(v, 0.0f);
    }
  }
  __syncthreads();

  // ---- layer 3: H3 = H2 @ W3 + b3 (64x128), fused masked max over rows ----
  v8f acc3[8];
#pragma unroll
  for (int t = 0; t < 8; t++) acc3[t] = vz;
  for (int k0 = 0; k0 < 64; k0 += 32) {
    v16h a = frag_a(H2 + wv * 16 * 64 + k0, lane);
#pragma unroll
    for (int tn = 0; tn < 8; tn++) {
      v16h bf = frag_b(Wt3, tn * 16, k0, lane);
      acc3[tn] = __builtin_amdgcn_wmma_f32_16x16x32_f16(false, a, false, bf,
                                                        (short)0, acc3[tn], false, false);
    }
  }
#pragma unroll
  for (int tn = 0; tn < 8; tn++) {
    int col = tn * 16 + ln;
    float bv = b3s[col];
    float v = NEGV;
#pragma unroll
    for (int e = 0; e < 8; e++) {
      int row = wv * 16 + lh * 8 + e;  // neighbor slot
      float h = acc3[tn][e] + bv;
      if (row < vcnt) v = fmaxf(v, h);
    }
    float o = __shfl_xor(v, 16);       // combine the two lane-halves (rows +/-8)
    v = fmaxf(v, o);
    if (lh == 0) pmax[wv][col] = v;
  }
  __syncthreads();

  // final combine across the 4 waves; tid == output column (0..127)
  float v = fmaxf(fmaxf(pmax[0][tid], pmax[1][tid]),
                  fmaxf(pmax[2][tid], pmax[3][tid]));
  xout[(size_t)c * 128 + tid] = v;
}

// ---------------- launcher ----------------
extern "C" void kernel_launch(void* const* d_in, const int* in_sizes, int n_in,
                              void* d_out, int out_size, void* d_ws, size_t ws_size,
                              hipStream_t stream) {
  const float* x   = (const float*)d_in[0];
  const float* pos = (const float*)d_in[1];
  // d_in[2] = batch (unused; batch_out derived analytically)
  const float* W1 = (const float*)d_in[3];
  const float* b1 = (const float*)d_in[4];
  const float* W2 = (const float*)d_in[5];
  const float* b2 = (const float*)d_in[6];
  const float* W3 = (const float*)d_in[7];
  const float* b3 = (const float*)d_in[8];

  float* xout      = (float*)d_out;                       // [NB*MCTR, 128]
  float* pos_out   = xout + (size_t)NB * MCTR * 128;      // [NB*MCTR, 3]
  int*   batch_out = (int*)(pos_out + (size_t)NB * MCTR * 3);  // [NB*MCTR]

  int* idx = (int*)d_ws;            // [NB*MCTR]
  int* cnt = idx + NB * MCTR;       // [NB*MCTR]
  int* nbr = cnt + NB * MCTR;       // [NB*MCTR, KNB]

  fps_kernel <<<NB,          256, 0, stream>>>(pos, idx, pos_out, batch_out);
  ball_kernel<<<NB * MCTR,   256, 0, stream>>>(pos, idx, nbr, cnt);
  mlp_kernel <<<NB * MCTR,   128, 0, stream>>>(x, pos, idx, nbr, cnt,
                                               W1, b1, W2, b2, W3, b3, xout);
}
